// MultiHeadedAttention_23321672417981
// MI455X (gfx1250) — compile-verified, matrix operand order checked by reading, not run on hardware
//
#include <hip/hip_runtime.h>

#define B_ 2
#define S_ 512
#define D_ 512
#define H_ 8
#define DK_ 64

typedef __attribute__((ext_vector_type(16))) _Float16 v16h;
typedef __attribute__((ext_vector_type(8)))  float    v8f;
typedef __attribute__((ext_vector_type(4)))  int      v4i;

#if defined(__gfx1250__) && __has_builtin(__builtin_amdgcn_global_load_async_to_lds_b128)
#define HAVE_ASYNC_LDS 1
#endif

#define AS1 __attribute__((address_space(1)))
#define AS3 __attribute__((address_space(3)))

// 16-byte global -> LDS copy; async DMA (ASYNCcnt) on gfx1250 when available.
__device__ inline void cp16_g2l(const float* g, float* l) {
#if defined(HAVE_ASYNC_LDS)
  __builtin_amdgcn_global_load_async_to_lds_b128((AS1 v4i*)g, (AS3 v4i*)l, 0, 0);
#else
  *(float4*)l = *(const float4*)g;
#endif
}
__device__ inline void cp_wait() {
#if defined(HAVE_ASYNC_LDS)
#if __has_builtin(__builtin_amdgcn_s_wait_asynccnt)
  __builtin_amdgcn_s_wait_asynccnt(0);
#else
  asm volatile("s_wait_asynccnt 0x0" ::: "memory");
#endif
#endif
}

// ---- monotone float<->u32 encoding for atomic min/max on floats ----
__device__ inline unsigned enc_f32(float f) {
  unsigned u = __float_as_uint(f);
  return (u & 0x80000000u) ? ~u : (u | 0x80000000u);
}
__device__ inline float dec_f32(unsigned u) {
  u = (u & 0x80000000u) ? (u & 0x7FFFFFFFu) : ~u;
  return __uint_as_float(u);
}

__global__ void init_mm(unsigned* mm) {
  mm[0] = 0xFFFFFFFFu;  // running min (encoded)
  mm[1] = 0u;           // running max (encoded)
}

__global__ void cvt_f32_f16(const float* __restrict__ src,
                            _Float16* __restrict__ dst, int n) {
  int i = blockIdx.x * blockDim.x + threadIdx.x;
  if (i < n) dst[i] = (_Float16)src[i];
}

// Y[m][n] = sum_k X[m][k] * W[n][k] + bias[n]
// X: f16 [M x K] row-major, W: f16 [N x K] row-major (== Wt of x@W.T)
// Each wave computes a 16x32 strip (2 accumulators, A-fragment reuse).
// mode 0: store f32 out[m*N+n]
// mode 1: head-split store to [B,H,S,DK] + global min/max atomics
// mode 2: head-split store, no atomics
__global__ __launch_bounds__(256) void gemm_xwt(
    const _Float16* __restrict__ X, const _Float16* __restrict__ W,
    const float* __restrict__ bias, float* __restrict__ out,
    int M, int N, int K, int mode, unsigned* __restrict__ mm) {
  const int lane = threadIdx.x & 31;
  const int wave = threadIdx.x >> 5;
  const int stripsN = N >> 5;                      // 32-wide N strips
  const int totalStrips = (M >> 4) * stripsN;
  const int strip = blockIdx.x * (blockDim.x >> 5) + wave;
  if (strip >= totalStrips) return;

  const int m0 = (strip / stripsN) << 4;
  const int n0 = (strip % stripsN) << 5;
  const int row  = lane & 15;          // M (for A) / N (for B) within tile
  const int koff = (lane >> 4) << 4;   // lanes 16..31 hold K=16..31 half

  const _Float16* ap = X + (size_t)(m0 + row) * K + koff;
  const _Float16* bp = W + (size_t)(n0 + row) * K + koff;

  v8f acc[2] = {};
  for (int k0 = 0; k0 < K; k0 += 32) {
    v16h a = *(const v16h*)(ap + k0);
#pragma unroll
    for (int t = 0; t < 2; ++t) {
      v16h b = *(const v16h*)(bp + (size_t)(t << 4) * K + k0);
      acc[t] = __builtin_amdgcn_wmma_f32_16x16x32_f16(false, a, false, b,
                                                      (short)0, acc[t], false, false);
    }
  }

  float lmin = 3.4e38f, lmax = -3.4e38f;
#pragma unroll
  for (int t = 0; t < 2; ++t) {
    const int nn = n0 + (t << 4) + (lane & 15);
    const float bn = bias[nn];
#pragma unroll
    for (int r = 0; r < 8; ++r) {
      const int m = m0 + r + ((lane >> 4) << 3);
      const float v = acc[t][r] + bn;
      if (mode == 0) {
        out[(size_t)m * N + nn] = v;
      } else {
        const int b_ = m >> 9, s = m & (S_ - 1);
        const int h = nn >> 6, dk = nn & (DK_ - 1);
        out[(((size_t)(b_ * H_ + h)) * S_ + s) * DK_ + dk] = v;
        lmin = fminf(lmin, v);
        lmax = fmaxf(lmax, v);
      }
    }
  }
  if (mode == 1) {
#pragma unroll
    for (int msk = 16; msk >= 1; msk >>= 1) {
      lmin = fminf(lmin, __shfl_xor(lmin, msk, 32));
      lmax = fmaxf(lmax, __shfl_xor(lmax, msk, 32));
    }
    if (lane == 0) {
      atomicMin(&mm[0], enc_f32(lmin));
      atomicMax(&mm[1], enc_f32(lmax));
    }
  }
}

// One block = 16 query rows of one (b,h). 128 threads (4 waves).
// Scores (L1 pairwise, VALU, LDS-tiled with async g->LDS DMA) -> softmax ->
// P@V via WMMA.
__global__ __launch_bounds__(128) void attn_kernel(
    const float* __restrict__ qws, const float* __restrict__ kws,
    const float* __restrict__ vws, _Float16* __restrict__ ctx,
    const unsigned* __restrict__ mm) {
  __shared__ __align__(16) float sQ[16 * DK_];     // 4 KB
  __shared__ float sS[16 * S_];                    // 32 KB
  __shared__ __align__(32) float sKP[64 * DK_];    // 16 KB: K-tile, reused as P(f16)
  __shared__ float sQC[16];
  __shared__ float sKC[64];

  const int tid = threadIdx.x;
  const int bid = blockIdx.x;
  const int qt = bid & ((S_ / 16) - 1);
  const int h  = (bid >> 5) & (H_ - 1);
  const int b  = bid >> 8;
  const int qbase = qt << 4;

  const float mn = dec_f32(mm[0]);
  const float mx = dec_f32(mm[1]);
  const float inv_range = 1.0f / (mx - mn);
  const float mid = 0.5f * (mx + mn);

  const float* Qh = qws + ((size_t)(b * H_ + h)) * S_ * DK_;
  const float* Kh = kws + ((size_t)(b * H_ + h)) * S_ * DK_;
  const float* Vh = vws + ((size_t)(b * H_ + h)) * S_ * DK_;

  // Q tile: 16x64 f32, staged via async-to-LDS (2 x b128 per thread)
  for (int idx = tid; idx < (16 * DK_) / 4; idx += 128)
    cp16_g2l(Qh + (size_t)qbase * DK_ + idx * 4, sQ + idx * 4);
  cp_wait();
  __syncthreads();
  if (tid < 16) {
    float s = 0.f;
    for (int d = 0; d < DK_; ++d) s += fabsf(sQ[tid * DK_ + d] - mid);
    sQC[tid] = s;
  }

  // ---- pairwise L1 scores, 64 keys per LDS tile ----
  for (int jb = 0; jb < S_; jb += 64) {
    __syncthreads();
    for (int idx = tid; idx < (64 * DK_) / 4; idx += 128)
      cp16_g2l(Kh + (size_t)jb * DK_ + idx * 4, sKP + idx * 4);
    cp_wait();
    __syncthreads();
    if (tid < 64) {
      float s = 0.f;
      for (int d = 0; d < DK_; ++d) s += fabsf(sKP[tid * DK_ + d] - mid);
      sKC[tid] = s;
    }
    __syncthreads();
    const int i = tid & 15;
    for (int jl = tid >> 4; jl < 64; jl += 8) {
      float s1 = 0.f;
#pragma unroll 8
      for (int d = 0; d < DK_; ++d)
        s1 += fabsf(sQ[i * DK_ + d] - sKP[jl * DK_ + d]);
      // (ALPHA/2)*(DK - L1/range) + (BETA/4)*(qc + kc)/range
      sS[i * S_ + jb + jl] = 0.15f * (64.f - s1 * inv_range) +
                             0.175f * (sQC[i] + sKC[jl]) * inv_range;
    }
  }
  __syncthreads();

  // ---- softmax per row, write P as f16 into reused sKP buffer ----
  _Float16* sP = (_Float16*)sKP;  // 16*512 f16 == 16 KB, K-tile done
  if (tid < 16) {
    float rmax = -3.4e38f;
    for (int j = 0; j < S_; ++j) rmax = fmaxf(rmax, sS[tid * S_ + j]);
    float sum = 0.f;
    for (int j = 0; j < S_; ++j) {
      float e = __expf(sS[tid * S_ + j] - rmax);
      sS[tid * S_ + j] = e;
      sum += e;
    }
    const float inv = 1.0f / sum;
    for (int j = 0; j < S_; ++j)
      sP[tid * S_ + j] = (_Float16)(sS[tid * S_ + j] * inv);
  }
  __syncthreads();

  // ---- P @ V with WMMA: 4 waves, one 16-wide dk stripe each ----
  const int lane  = tid & 31;
  const int wave  = tid >> 5;           // 0..3
  const int n0    = wave << 4;          // dk tile base
  const int col   = lane & 15;
  const int khalf = (lane >> 4) << 4;
  v8f c = {};
  for (int kb = 0; kb < S_; kb += 32) {
    v16h a = *(const v16h*)(sP + col * S_ + kb + khalf);  // row = lane&15
    v16h bfr;
#pragma unroll
    for (int t = 0; t < 16; ++t)
      bfr[t] = (_Float16)Vh[(size_t)(kb + khalf + t) * DK_ + n0 + col];
    c = __builtin_amdgcn_wmma_f32_16x16x32_f16(false, a, false, bfr,
                                               (short)0, c, false, false);
  }
#pragma unroll
  for (int r = 0; r < 8; ++r) {
    const int irow = r + ((lane >> 4) << 3);
    const int s = qbase + irow;
    const int d = h * DK_ + n0 + col;
    ctx[((size_t)(b * S_ + s)) * D_ + d] = (_Float16)c[r];
  }
}

extern "C" void kernel_launch(void* const* d_in, const int* in_sizes, int n_in,
                              void* d_out, int out_size, void* d_ws, size_t ws_size,
                              hipStream_t stream) {
  const float* query = (const float*)d_in[0];
  const float* key   = (const float*)d_in[1];
  const float* value = (const float*)d_in[2];
  const float* Wq = (const float*)d_in[3];
  const float* bq = (const float*)d_in[4];
  const float* Wk = (const float*)d_in[5];
  const float* bk = (const float*)d_in[6];
  const float* Wv = (const float*)d_in[7];
  const float* bv = (const float*)d_in[8];
  const float* Wo = (const float*)d_in[9];
  const float* bo = (const float*)d_in[10];

  const int M = B_ * S_;             // 1024
  const size_t ACT = (size_t)M * D_; // 524288
  const size_t WEL = (size_t)D_ * D_;

  // ---- workspace carve-up ----
  char* ws = (char*)d_ws;
  _Float16* xq = (_Float16*)ws;              ws += ACT * 2;
  _Float16* xk = (_Float16*)ws;              ws += ACT * 2;
  _Float16* xv = (_Float16*)ws;              ws += ACT * 2;
  _Float16* wqh = (_Float16*)ws;             ws += WEL * 2;
  _Float16* wkh = (_Float16*)ws;             ws += WEL * 2;
  _Float16* wvh = (_Float16*)ws;             ws += WEL * 2;
  _Float16* woh = (_Float16*)ws;             ws += WEL * 2;
  float* qws = (float*)ws;                   ws += ACT * 4;
  float* kws = (float*)ws;                   ws += ACT * 4;
  float* vws = (float*)ws;                   ws += ACT * 4;
  _Float16* ctx = (_Float16*)ws;             ws += ACT * 2;
  unsigned* mm = (unsigned*)ws;

  init_mm<<<1, 1, 0, stream>>>(mm);

  const int CT = 256;
  cvt_f32_f16<<<(int)((ACT + CT - 1) / CT), CT, 0, stream>>>(query, xq, (int)ACT);
  cvt_f32_f16<<<(int)((ACT + CT - 1) / CT), CT, 0, stream>>>(key,   xk, (int)ACT);
  cvt_f32_f16<<<(int)((ACT + CT - 1) / CT), CT, 0, stream>>>(value, xv, (int)ACT);
  cvt_f32_f16<<<(int)((WEL + CT - 1) / CT), CT, 0, stream>>>(Wq, wqh, (int)WEL);
  cvt_f32_f16<<<(int)((WEL + CT - 1) / CT), CT, 0, stream>>>(Wk, wkh, (int)WEL);
  cvt_f32_f16<<<(int)((WEL + CT - 1) / CT), CT, 0, stream>>>(Wv, wvh, (int)WEL);
  cvt_f32_f16<<<(int)((WEL + CT - 1) / CT), CT, 0, stream>>>(Wo, woh, (int)WEL);

  const int strips = (M / 16) * (D_ / 32);  // 1024 waves
  const int gblocks = strips / 8;           // 8 waves per 256-thread block
  gemm_xwt<<<gblocks, 256, 0, stream>>>(xq, wqh, bq, qws, M, D_, D_, 1, mm);
  gemm_xwt<<<gblocks, 256, 0, stream>>>(xk, wkh, bk, kws, M, D_, D_, 1, mm);
  gemm_xwt<<<gblocks, 256, 0, stream>>>(xv, wvh, bv, vws, M, D_, D_, 2, mm);

  attn_kernel<<<B_ * H_ * (S_ / 16), 128, 0, stream>>>(qws, kws, vws, ctx, mm);

  gemm_xwt<<<gblocks, 256, 0, stream>>>(ctx, woh, bo, (float*)d_out, M, D_, D_, 0, mm);
}
